// MoE_81423989997720
// MI455X (gfx1250) — matrix operand order repaired and checked by reading, not compiled, not run on hardware
//
#include <hip/hip_runtime.h>
#include <cstdint>

// ---------------------------------------------------------------------------
// MoE matting head for MI455X (gfx1250, wave32, WMMA).
// Pipeline: downsample -> gating GEMV -> top2 gates/loss -> sparse expert
// conv as implicit GEMM via v_wmma_f32_16x16x32_f16 -> exp/combine/store.
// Round 3: hard-cap conv kernel at 256 VGPRs (amdgpu_num_vgpr) so the
// allocator rematerializes the LDS A-tile loads instead of spilling into
// MSB-addressed VGPRs (s_set_vgpr_msb) -> better occupancy, cleaner SALU.
// ---------------------------------------------------------------------------

typedef __attribute__((ext_vector_type(16))) _Float16 v16h;
typedef __attribute__((ext_vector_type(8)))  float    v8f;

#define NB 8            // batch
#define NE 8            // experts
#define COUT 37
#define HS 256
#define WSP 256
#define HWS (HS * WSP)              // 65536
#define XCH 6
#define XSIZE (XCH * HWS)           // 393216 per batch
#define KDIM 54                     // 6*3*3
#define EPS_COMB 2.2204460492503131e-16f

// output section offsets (flattened tuple: pha, fgr, err, hid, loss)
#define OFF_PHA 0
#define OFF_FGR 524288
#define OFF_ERR 2097152
#define OFF_HID 2621440
#define OFF_LOSS 19398656

// ---------------------------------------------------------------------------
__global__ void init_logits(float* logits) {
    int t = threadIdx.x;
    if (t < NB * NE) logits[t] = 0.0f;
}

// ---------------------------------------------------------------------------
// Bilinear 4x downsample: out = avg of rows 4h+1,4h+2 x cols 4w+1,4w+2.
// float4 loads keep HBM traffic at the two needed rows only.
__global__ __launch_bounds__(256) void downsample_kernel(
    const float* __restrict__ src, const float* __restrict__ bgr,
    float* __restrict__ xf, _Float16* __restrict__ xh) {
    size_t idx = (size_t)blockIdx.x * 256 + threadIdx.x;   // B*6*256*256
    int wo = idx & 255;
    int ho = (idx >> 8) & 255;
    int ci = (int)((idx >> 16) % 6);
    int b  = (int)(idx / (size_t)XSIZE);
    const float* in = (ci < 3) ? src : bgr;
    int cc = (ci < 3) ? ci : ci - 3;
    const float* p = in + ((size_t)b * 3 + cc) * 1024 * 1024;
    int hi4 = 4 * ho + 1, wi4 = 4 * wo;
    float4 r1 = *(const float4*)(p + (size_t)hi4 * 1024 + wi4);
    float4 r2 = *(const float4*)(p + (size_t)(hi4 + 1) * 1024 + wi4);
    float v = 0.25f * (r1.y + r1.z + r2.y + r2.z);
    xf[idx] = v;
    xh[idx] = (_Float16)v;
}

// ---------------------------------------------------------------------------
// logits[b][e] += sum_i x[b][i] * w_gate[i][e]  (chunked, atomic combine)
__global__ __launch_bounds__(256) void gate_gemm_kernel(
    const float* __restrict__ xf, const float* __restrict__ wg,
    float* __restrict__ logits) {
    int b = blockIdx.y;
    int chunk = blockIdx.x;                 // 8 chunks
    int tid = threadIdx.x;
    const int CL = XSIZE / 8;               // 49152
    float acc[NE];
#pragma unroll
    for (int e = 0; e < NE; ++e) acc[e] = 0.0f;
    const float* xb = xf + (size_t)b * XSIZE;
    for (int i = chunk * CL + tid; i < (chunk + 1) * CL; i += 256) {
        float xv = xb[i];
        const float* wr = wg + (size_t)i * NE;
#pragma unroll
        for (int e = 0; e < NE; ++e) acc[e] += xv * wr[e];
    }
    __shared__ float red[256];
#pragma unroll
    for (int e = 0; e < NE; ++e) {
        red[tid] = acc[e];
        __syncthreads();
        for (int off = 128; off > 0; off >>= 1) {
            if (tid < off) red[tid] += red[tid + off];
            __syncthreads();
        }
        if (tid == 0) atomicAdd(&logits[b * NE + e], red[0]);
        __syncthreads();
    }
}

// ---------------------------------------------------------------------------
__device__ inline float cv_squared8(const float* v) {
    float s = 0.0f;
#pragma unroll
    for (int i = 0; i < 8; ++i) s += v[i];
    float mean = s * 0.125f;
    float var = 0.0f;
#pragma unroll
    for (int i = 0; i < 8; ++i) { float d = v[i] - mean; var += d * d; }
    var *= (1.0f / 7.0f);                   // ddof=1
    return var / (mean * mean + 1e-10f);
}

__global__ void gate_finalize_kernel(const float* __restrict__ logits,
                                     int* __restrict__ gidx,
                                     float* __restrict__ gval,
                                     float* __restrict__ loss_out) {
    if (threadIdx.x != 0) return;
    float importance[NE], load[NE];
#pragma unroll
    for (int e = 0; e < NE; ++e) { importance[e] = 0.0f; load[e] = 0.0f; }
    for (int b = 0; b < NB; ++b) {
        const float* L = logits + b * NE;
        int i0 = 0, i1 = -1;
        float v0 = L[0], v1 = -3.4e38f;
        for (int e = 1; e < NE; ++e) {
            float v = L[e];
            if (v > v0) { v1 = v0; i1 = i0; v0 = v; i0 = e; }
            else if (v > v1) { v1 = v; i1 = e; }
        }
        float ex0 = __expf(v0 - v0), ex1 = __expf(v1 - v0);
        float inv = 1.0f / (ex0 + ex1);
        float g0 = ex0 * inv, g1 = ex1 * inv;
        gidx[b * 2 + 0] = i0; gidx[b * 2 + 1] = i1;
        gval[b * 2 + 0] = g0; gval[b * 2 + 1] = g1;
        importance[i0] += g0; importance[i1] += g1;
        load[i0] += 1.0f; load[i1] += 1.0f;
    }
    *loss_out = 0.01f * (cv_squared8(importance) + cv_squared8(load));
}

// ---------------------------------------------------------------------------
// Output scatter into the concatenated (pha,fgr,err,hid) sections.
__device__ inline size_t out_off(int b, int c, int h, int w) {
    size_t hw = (size_t)h * WSP + w;
    if (c == 0)      return (size_t)OFF_PHA + (size_t)b * HWS + hw;
    else if (c <= 3) return (size_t)OFF_FGR + ((size_t)(b * 3 + (c - 1))) * HWS + hw;
    else if (c == 4) return (size_t)OFF_ERR + (size_t)b * HWS + hw;
    else             return (size_t)OFF_HID + ((size_t)(b * 32 + (c - 5))) * HWS + hw;
}

// ---------------------------------------------------------------------------
// Sparse MoE conv: per block = (4 rows, one batch). Implicit GEMM
// M=48(pad of 37) x K=64(pad of 54) x N=16-pixel groups, both top-2 experts.
// A (weights) pre-packed in LDS in the 16-bit A-matrix WMMA layout and
// refetched per group with ds_load_b128; B (im2col) gathered from an LDS
// halo strip. amdgpu_num_vgpr(256) makes the allocator rematerialize the
// A-tile LDS loads under pressure instead of using MSB-addressed VGPRs.
#define LDSW 260        // 258 used cols (w = -1..256) + pad
#define ATILES 12       // 2 experts x 3 m-tiles x 2 k-tiles
__global__ __launch_bounds__(128)
__attribute__((amdgpu_num_vgpr(256)))
void conv_combine_kernel(
    const _Float16* __restrict__ xh, const float* __restrict__ ew,
    const float* __restrict__ eb, const int* __restrict__ gidx,
    const float* __restrict__ gval, float* __restrict__ out) {
    __shared__ __align__(32) _Float16 xs[XCH * 6 * LDSW];   // 18,720 B
    __shared__ __align__(32) _Float16 as[ATILES * 512];     // 12,288 B

    int b  = blockIdx.y;
    int h0 = blockIdx.x * 4;
    int tid  = threadIdx.x;
    int wave = tid >> 5;
    int lane = tid & 31;
    int lm = lane & 15;
    bool hiL = lane >= 16;

    int e_sel[2];
    float g_sel[2];
    e_sel[0] = gidx[b * 2 + 0]; e_sel[1] = gidx[b * 2 + 1];
    g_sel[0] = gval[b * 2 + 0]; g_sel[1] = gval[b * 2 + 1];

    // ---- stage 6 input rows (h0-1 .. h0+4) x 6 channels into LDS as f16 ----
    for (int idx = tid; idx < XCH * 6 * LDSW; idx += 128) {
        int col = idx % LDSW;
        int t = idx / LDSW;
        int r = t % 6;
        int ci = t / 6;
        int h = h0 - 1 + r;
        int w = col - 1;
        _Float16 v = (_Float16)0.0f;
        if (h >= 0 && h < HS && w >= 0 && w < WSP)
            v = xh[(((size_t)b * XCH + ci) * HS + h) * WSP + w];
        xs[idx] = v;
    }

    // ---- stage A tiles into LDS in WMMA 16-bit A layout --------------------
    // tile = ((s*3 + mt)*2 + kt); half index = tile*512 + lane*16 + j
    // lanes 0-15 row M=lane: element j -> K = j + (j>=8 ? 8 : 0) (+8 if hi)
    for (int idx = tid; idx < ATILES * 512; idx += 128) {
        int tile = idx >> 9;
        int within = idx & 511;
        int l = within >> 4;
        int j = within & 15;
        int s  = tile / 6;
        int r6 = tile % 6;
        int mt = r6 >> 1;
        int kt = r6 & 1;
        int c = mt * 16 + (l & 15);
        int K = kt * 32 + j + (j >= 8 ? 8 : 0) + ((l >= 16) ? 8 : 0);
        float v = 0.0f;
        if (c < COUT && K < KDIM)
            v = ew[((size_t)e_sel[s] * COUT + c) * KDIM + K];
        as[idx] = (_Float16)v;
    }
    __syncthreads();

    // ---- fold gate & bias: scale = g * exp(bias[c]) -----------------------
    float sc[2][3][8];
#pragma unroll
    for (int s = 0; s < 2; ++s)
#pragma unroll
        for (int mt = 0; mt < 3; ++mt)
#pragma unroll
            for (int v = 0; v < 8; ++v) {
                int c = mt * 16 + v + (hiL ? 8 : 0);
                sc[s][mt][v] = (c < COUT)
                    ? g_sel[s] * __expf(eb[e_sel[s] * COUT + c]) : 0.0f;
            }

    int h = h0 + wave;   // this wave's output row; LDS rows r = wave + ky
    const v16h* Ald = (const v16h*)as;      // 32B per lane per tile

    // ---- sweep 16 pixel-groups of 16 along the row ------------------------
    for (int g = 0; g < 16; ++g) {
        int w0 = g * 16;
        // B tiles (32x16): element j -> K = j + (j>=8 ? 8 : 0) (+8 if hi), matching A
        v16h Bt[2];
#pragma unroll
        for (int kt = 0; kt < 2; ++kt) {
            v16h bb;
#pragma unroll
            for (int j = 0; j < 16; ++j) {
                int K = kt * 32 + (hiL ? 8 : 0) + j + (j >= 8 ? 8 : 0);
                _Float16 v = (_Float16)0.0f;
                if (K < KDIM) {
                    int ci = K / 9, rr = K % 9, ky = rr / 3, kx = rr % 3;
                    v = xs[(ci * 6 + wave + ky) * LDSW + (w0 + lm + kx)];
                }
                bb[j] = v;
            }
            Bt[kt] = bb;
        }

        float outv[3][8];
#pragma unroll
        for (int s = 0; s < 2; ++s) {
            v8f acc[3];
#pragma unroll
            for (int mt = 0; mt < 3; ++mt)
                acc[mt] = (v8f){0, 0, 0, 0, 0, 0, 0, 0};
#pragma unroll
            for (int mt = 0; mt < 3; ++mt) {
                v16h a0 = Ald[((s * 3 + mt) * 2 + 0) * 32 + lane];
                v16h a1 = Ald[((s * 3 + mt) * 2 + 1) * 32 + lane];
                acc[mt] = __builtin_amdgcn_wmma_f32_16x16x32_f16(
                    false, a0, false, Bt[0], (short)0, acc[mt], false, false);
                acc[mt] = __builtin_amdgcn_wmma_f32_16x16x32_f16(
                    false, a1, false, Bt[1], (short)0, acc[mt], false, false);
            }
            // partial combine: outv (+)= g_s * exp(conv_s + bias_s)
#pragma unroll
            for (int mt = 0; mt < 3; ++mt)
#pragma unroll
                for (int v = 0; v < 8; ++v) {
                    float t = sc[s][mt][v] * __expf(acc[mt][v]);
                    outv[mt][v] = (s == 0) ? t : (outv[mt][v] + t);
                }
        }

        // EPS rule + scatter into concatenated output sections
#pragma unroll
        for (int mt = 0; mt < 3; ++mt)
#pragma unroll
            for (int v = 0; v < 8; ++v) {
                int c = mt * 16 + v + (hiL ? 8 : 0);
                if (c < COUT) {
                    float val = outv[mt][v];
                    if (val == 0.0f) val = EPS_COMB;
                    out[out_off(b, c, h, w0 + lm)] = val;
                }
            }
    }
}

// ---------------------------------------------------------------------------
extern "C" void kernel_launch(void* const* d_in, const int* in_sizes, int n_in,
                              void* d_out, int out_size, void* d_ws, size_t ws_size,
                              hipStream_t stream) {
    const float* src      = (const float*)d_in[0];
    const float* bgr      = (const float*)d_in[1];
    const float* w_gate   = (const float*)d_in[2];
    const float* expert_w = (const float*)d_in[3];
    const float* expert_b = (const float*)d_in[4];
    float* out = (float*)d_out;

    // workspace carve
    char* ws = (char*)d_ws;
    float*    x_f32  = (float*)ws;                            // 12,582,912 B
    _Float16* x_f16  = (_Float16*)(ws + 12582912);            //  6,291,456 B
    float*    logits = (float*)(ws + 18874368);               //        256 B
    int*      gidx   = (int*)(ws + 18874624);                 //         64 B
    float*    gvalp  = (float*)(ws + 18874688);               //         64 B

    init_logits<<<1, 64, 0, stream>>>(logits);
    downsample_kernel<<<(NB * XSIZE) / 256, 256, 0, stream>>>(src, bgr, x_f32, x_f16);
    gate_gemm_kernel<<<dim3(8, NB), 256, 0, stream>>>(x_f32, w_gate, logits);
    gate_finalize_kernel<<<1, 32, 0, stream>>>(logits, gidx, gvalp, out + OFF_LOSS);
    conv_combine_kernel<<<dim3(HS / 4, NB), 128, 0, stream>>>(
        x_f16, expert_w, expert_b, gidx, gvalp, out);
}
